// GAT_20950850470457
// MI455X (gfx1250) — compile-verified, matrix operand order checked by reading, not run on hardware
//
#include <hip/hip_runtime.h>
#include <hip/hip_bf16.h>

#define N_NODES 50000
#define N_EDGES 1600000
#define IN_F    256
#define HID     32
#define HEADS   4
#define F2      128        // HEADS*HID
#define NC      16
#define NEG_SLOPE 0.2f

typedef __attribute__((ext_vector_type(16))) _Float16 v16h;
typedef __attribute__((ext_vector_type(8)))  _Float16 v8h;
typedef __attribute__((ext_vector_type(8)))  float    v8f;

// ---------- helpers ----------
__device__ __forceinline__ unsigned enc_key(float f) {
    unsigned b = __float_as_uint(f);
    return (b & 0x80000000u) ? ~b : (b | 0x80000000u);
}
__device__ __forceinline__ float dec_key(unsigned k) {
    unsigned b = (k & 0x80000000u) ? (k ^ 0x80000000u) : ~k;
    // empty segment / non-finite -> 0 (matches reference isfinite masking)
    if ((b & 0x7f800000u) == 0x7f800000u) return 0.0f;
    return __uint_as_float(b);
}
__device__ __forceinline__ float leaky(float v) { return v > 0.0f ? v : NEG_SLOPE * v; }

// ---------- precision conversion / weight transpose ----------
__global__ void k_f32_to_f16(const float* __restrict__ in, _Float16* __restrict__ out, int n) {
    int i = blockIdx.x * blockDim.x + threadIdx.x;
    if (i < n) out[i] = (_Float16)in[i];
}
// in: [R][C] row-major f32  ->  out: [C][R] f16
__global__ void k_transpose_f16(const float* __restrict__ in, _Float16* __restrict__ out,
                                int R, int C) {
    int i = blockIdx.x * blockDim.x + threadIdx.x;
    if (i >= R * C) return;
    int r = i / C, c = i % C;
    out[c * R + r] = (_Float16)in[i];
}

// ---------- WMMA GEMM: C[M][Ncol] = A[M][K](f16) * Bt[Ncol][K](f16), f32 accum ----------
// One wave per 16x16 output tile. M % 16 == 0 (50000 = 3125*16). EXEC all-ones per wave.
template <int K>
__global__ void __launch_bounds__(256) k_gemm_wmma(const _Float16* __restrict__ A,
                                                   const _Float16* __restrict__ Bt,
                                                   float* __restrict__ C,
                                                   int M, int Ncol) {
    int wave = (blockIdx.x * blockDim.x + threadIdx.x) >> 5;
    int lane = threadIdx.x & 31;
    int mTiles = M >> 4, nTiles = Ncol >> 4;
    if (wave >= mTiles * nTiles) return;
    int mt = wave / nTiles, nt = wave % nTiles;
    int half = lane >> 4;      // 0: K chunk {0..7,16..23}; 1: {8..15,24..31}
    int idx  = lane & 15;

    const _Float16* arow = A  + (size_t)(mt * 16 + idx) * K + half * 8;
    const _Float16* brow = Bt + (size_t)(nt * 16 + idx) * K + half * 8;

    v8f acc = {};
#pragma unroll
    for (int k0 = 0; k0 < K; k0 += 32) {
        union { v16h v; struct { v8h lo, hi; } h; } a, b;
        a.h.lo = *(const v8h*)(arow + k0);
        a.h.hi = *(const v8h*)(arow + k0 + 16);
        b.h.lo = *(const v8h*)(brow + k0);
        b.h.hi = *(const v8h*)(brow + k0 + 16);
        acc = __builtin_amdgcn_wmma_f32_16x16x32_f16(false, a.v, false, b.v,
                                                     (short)0, acc, false, false);
    }
    // C/D layout: VGPR r -> M = r + half*8 ; N = lane&15
    float* crow = C + (size_t)(mt * 16 + half * 8) * Ncol + nt * 16 + idx;
#pragma unroll
    for (int r = 0; r < 8; ++r) crow[(size_t)r * Ncol] = acc[r];
}

// ---------- attention projections el/er : one thread per (node, head) ----------
__global__ void k_el_er(const float* __restrict__ feat, const float* __restrict__ al,
                        const float* __restrict__ ar, float* __restrict__ el,
                        float* __restrict__ er, int n, int H, int D) {
    int i = blockIdx.x * blockDim.x + threadIdx.x;
    if (i >= n * H) return;
    int node = i / H, h = i % H;
    const float* f = feat + (size_t)node * H * D + h * D;
    const float* a = al + h * D;
    const float* b = ar + h * D;
    float sl = 0.0f, sr = 0.0f;
    for (int d = 0; d < D; ++d) { float v = f[d]; sl += v * a[d]; sr += v * b[d]; }
    el[i] = sl; er[i] = sr;
}

// ---------- layer 1 edge passes (H = 4) ----------
__global__ void k_edge_max4(const int* __restrict__ src, const int* __restrict__ dst,
                            const float* __restrict__ el, const float* __restrict__ er,
                            unsigned* __restrict__ mx) {
    int e = blockIdx.x * blockDim.x + threadIdx.x;
    if (e >= N_EDGES) return;
    int s = src[e], d = dst[e];
    float4 a = *(const float4*)(el + 4 * (size_t)s);
    float4 b = *(const float4*)(er + 4 * (size_t)d);
    float v[4] = { a.x + b.x, a.y + b.y, a.z + b.z, a.w + b.w };
#pragma unroll
    for (int h = 0; h < 4; ++h)
        atomicMax(mx + 4 * (size_t)d + h, enc_key(leaky(v[h])));
}

__global__ void k_edge_exp4(const int* __restrict__ src, const int* __restrict__ dst,
                            const float* __restrict__ el, const float* __restrict__ er,
                            const unsigned* __restrict__ mx, float* __restrict__ ex,
                            float* __restrict__ den) {
    int e = blockIdx.x * blockDim.x + threadIdx.x;
    if (e >= N_EDGES) return;
    int s = src[e], d = dst[e];
    float4 a = *(const float4*)(el + 4 * (size_t)s);
    float4 b = *(const float4*)(er + 4 * (size_t)d);
    float v[4] = { a.x + b.x, a.y + b.y, a.z + b.z, a.w + b.w };
#pragma unroll
    for (int h = 0; h < 4; ++h) {
        float m = dec_key(mx[4 * (size_t)d + h]);
        float x = __expf(leaky(v[h]) - m);
        ex[4 * (size_t)e + h] = x;
        atomicAdd(den + 4 * (size_t)d + h, x);
    }
}

// one wave per edge: 32 lanes x float4 = 128 features; lane>>3 selects head
__global__ void __launch_bounds__(256) k_edge_agg4(const int* __restrict__ src,
                                                   const int* __restrict__ dst,
                                                   const float* __restrict__ feat,
                                                   const float* __restrict__ ex,
                                                   const float* __restrict__ den,
                                                   float* __restrict__ agg) {
    int wave = (blockIdx.x * blockDim.x + threadIdx.x) >> 5;
    int lane = threadIdx.x & 31;
    if (wave >= N_EDGES) return;
    int s = src[wave], d = dst[wave];
    int h = lane >> 3;
    float dn = den[4 * (size_t)d + h];
    float alpha = ex[4 * (size_t)wave + h] / fmaxf(dn, 1e-9f);
    float4 f = *(const float4*)(feat + (size_t)s * F2 + lane * 4);
    float* o = agg + (size_t)d * F2 + lane * 4;
    atomicAdd(o + 0, f.x * alpha);
    atomicAdd(o + 1, f.y * alpha);
    atomicAdd(o + 2, f.z * alpha);
    atomicAdd(o + 3, f.w * alpha);
}

// agg1 + b1 -> ELU -> f16 hidden
__global__ void k_fin1(const float* __restrict__ agg, const float* __restrict__ b1,
                       _Float16* __restrict__ h16) {
    int i = blockIdx.x * blockDim.x + threadIdx.x;
    if (i >= N_NODES * F2) return;
    float v = agg[i] + b1[i & (F2 - 1)];
    v = v > 0.0f ? v : (__expf(v) - 1.0f);
    h16[i] = (_Float16)v;
}

// ---------- layer 2 edge passes (H = 1, C = 16) ----------
__global__ void k_edge_max1(const int* __restrict__ src, const int* __restrict__ dst,
                            const float* __restrict__ el, const float* __restrict__ er,
                            unsigned* __restrict__ mx) {
    int e = blockIdx.x * blockDim.x + threadIdx.x;
    if (e >= N_EDGES) return;
    int s = src[e], d = dst[e];
    atomicMax(mx + d, enc_key(leaky(el[s] + er[d])));
}

__global__ void k_edge_exp1(const int* __restrict__ src, const int* __restrict__ dst,
                            const float* __restrict__ el, const float* __restrict__ er,
                            const unsigned* __restrict__ mx, float* __restrict__ ex,
                            float* __restrict__ den) {
    int e = blockIdx.x * blockDim.x + threadIdx.x;
    if (e >= N_EDGES) return;
    int s = src[e], d = dst[e];
    float x = __expf(leaky(el[s] + er[d]) - dec_key(mx[d]));
    ex[e] = x;
    atomicAdd(den + d, x);
}

// 4 threads per edge, float4 each (16 features)
__global__ void k_edge_agg1(const int* __restrict__ src, const int* __restrict__ dst,
                            const float* __restrict__ feat, const float* __restrict__ ex,
                            const float* __restrict__ den, float* __restrict__ agg) {
    int t = blockIdx.x * blockDim.x + threadIdx.x;
    if (t >= N_EDGES * 4) return;
    int e = t >> 2, q = t & 3;
    int s = src[e], d = dst[e];
    float alpha = ex[e] / fmaxf(den[d], 1e-9f);
    float4 f = *(const float4*)(feat + (size_t)s * NC + q * 4);
    float* o = agg + (size_t)d * NC + q * 4;
    atomicAdd(o + 0, f.x * alpha);
    atomicAdd(o + 1, f.y * alpha);
    atomicAdd(o + 2, f.z * alpha);
    atomicAdd(o + 3, f.w * alpha);
}

__global__ void k_fin2(const float* __restrict__ agg, const float* __restrict__ b2,
                       float* __restrict__ out) {
    int i = blockIdx.x * blockDim.x + threadIdx.x;
    if (i >= N_NODES * NC) return;
    out[i] = agg[i] + b2[i & (NC - 1)];
}

// ---------- launcher ----------
extern "C" void kernel_launch(void* const* d_in, const int* in_sizes, int n_in,
                              void* d_out, int out_size, void* d_ws, size_t ws_size,
                              hipStream_t stream) {
    const float* x   = (const float*)d_in[0];
    const int*   src = (const int*)  d_in[1];
    const int*   dst = (const int*)  d_in[2];
    const float* W1  = (const float*)d_in[3];
    const float* b1  = (const float*)d_in[4];
    const float* al1 = (const float*)d_in[5];
    const float* ar1 = (const float*)d_in[6];
    const float* W2  = (const float*)d_in[7];
    const float* b2  = (const float*)d_in[8];
    const float* al2 = (const float*)d_in[9];
    const float* ar2 = (const float*)d_in[10];
    float* out = (float*)d_out;
    (void)in_sizes; (void)n_in; (void)out_size; (void)ws_size;

    char* p = (char*)d_ws;
    auto carve = [&](size_t bytes) -> char* {
        char* r = p; p += (bytes + 255) & ~(size_t)255; return r;
    };
    _Float16* xh    = (_Float16*)carve((size_t)N_NODES * IN_F * 2);
    _Float16* w1t   = (_Float16*)carve((size_t)IN_F * F2 * 2);
    float*    feat1 = (float*)   carve((size_t)N_NODES * F2 * 4);
    float*    el1   = (float*)   carve((size_t)N_NODES * HEADS * 4);
    float*    er1   = (float*)   carve((size_t)N_NODES * HEADS * 4);
    unsigned* mx1   = (unsigned*)carve((size_t)N_NODES * HEADS * 4);
    float*    den1  = (float*)   carve((size_t)N_NODES * HEADS * 4);
    float*    ex1   = (float*)   carve((size_t)N_EDGES * HEADS * 4);
    float*    agg1  = (float*)   carve((size_t)N_NODES * F2 * 4);
    _Float16* h16   = (_Float16*)carve((size_t)N_NODES * F2 * 2);
    _Float16* w2t   = (_Float16*)carve((size_t)F2 * NC * 2);
    float*    feat2 = (float*)   carve((size_t)N_NODES * NC * 4);
    float*    el2   = (float*)   carve((size_t)N_NODES * 4);
    float*    er2   = (float*)   carve((size_t)N_NODES * 4);
    unsigned* mx2   = (unsigned*)carve((size_t)N_NODES * 4);
    float*    den2  = (float*)   carve((size_t)N_NODES * 4);
    float*    ex2   = (float*)   carve((size_t)N_EDGES * 4);
    float*    agg2  = (float*)   carve((size_t)N_NODES * NC * 4);

    // graph-capture-safe zero init of atomic targets (0x0 key == "-inf" sentinel)
    hipMemsetAsync(mx1,  0, (size_t)N_NODES * HEADS * 4, stream);
    hipMemsetAsync(den1, 0, (size_t)N_NODES * HEADS * 4, stream);
    hipMemsetAsync(agg1, 0, (size_t)N_NODES * F2 * 4,    stream);
    hipMemsetAsync(mx2,  0, (size_t)N_NODES * 4,         stream);
    hipMemsetAsync(den2, 0, (size_t)N_NODES * 4,         stream);
    hipMemsetAsync(agg2, 0, (size_t)N_NODES * NC * 4,    stream);

    const int T = 256;
    auto blk = [](long long n, int t) { return (int)((n + t - 1) / t); };

    // ----- layer 1 -----
    k_f32_to_f16   <<<blk((long long)N_NODES * IN_F, T), T, 0, stream>>>(x, xh, N_NODES * IN_F);
    k_transpose_f16<<<blk((long long)IN_F * F2, T), T, 0, stream>>>(W1, w1t, IN_F, F2);

    {   // 3125 * 8 tiles, 8 waves/block
        int waves = (N_NODES / 16) * (F2 / 16);
        k_gemm_wmma<IN_F><<<blk(waves, 8), T, 0, stream>>>(xh, w1t, feat1, N_NODES, F2);
    }
    k_el_er<<<blk((long long)N_NODES * HEADS, T), T, 0, stream>>>(feat1, al1, ar1, el1, er1,
                                                                  N_NODES, HEADS, HID);
    k_edge_max4<<<blk(N_EDGES, T), T, 0, stream>>>(src, dst, el1, er1, mx1);
    k_edge_exp4<<<blk(N_EDGES, T), T, 0, stream>>>(src, dst, el1, er1, mx1, ex1, den1);
    k_edge_agg4<<<blk((long long)N_EDGES * 32, T), T, 0, stream>>>(src, dst, feat1, ex1, den1, agg1);
    k_fin1<<<blk((long long)N_NODES * F2, T), T, 0, stream>>>(agg1, b1, h16);

    // ----- layer 2 -----
    k_transpose_f16<<<blk((long long)F2 * NC, T), T, 0, stream>>>(W2, w2t, F2, NC);
    {   // 3125 * 1 tiles
        int waves = (N_NODES / 16) * (NC / 16);
        k_gemm_wmma<F2><<<blk(waves, 8), T, 0, stream>>>(h16, w2t, feat2, N_NODES, NC);
    }
    k_el_er<<<blk(N_NODES, T), T, 0, stream>>>(feat2, al2, ar2, el2, er2, N_NODES, 1, NC);
    k_edge_max1<<<blk(N_EDGES, T), T, 0, stream>>>(src, dst, el2, er2, mx2);
    k_edge_exp1<<<blk(N_EDGES, T), T, 0, stream>>>(src, dst, el2, er2, mx2, ex2, den2);
    k_edge_agg1<<<blk((long long)N_EDGES * 4, T), T, 0, stream>>>(src, dst, feat2, ex2, den2, agg2);
    k_fin2<<<blk((long long)N_NODES * NC, T), T, 0, stream>>>(agg2, b2, out);
}
